// SelfAttentionLayer_73065983640152
// MI455X (gfx1250) — matrix operand order repaired and checked by reading, not compiled, not run on hardware
//
#include <hip/hip_runtime.h>
#include <hip/hip_bf16.h>
#include <hip/hip_fp16.h>

typedef __attribute__((ext_vector_type(16))) _Float16 v16h;
typedef __attribute__((ext_vector_type(8)))  float    v8f;

union AFrag { v16h v; _Float16 e[16]; };
union CFrag { v8f  v; float    e[8];  };

// ---- WMMA helpers: CDNA5 16x16x32 f16 layouts (ISA 7.12.2) ----------------

// A-matrix 16x32 f16: lane L holds row m=L%16; half=L/16 selects K runs
// [8*half, 8*half+8) and [16+8*half, 16+8*half+8).
__device__ __forceinline__ v16h load_a_f16(const _Float16* base, int stride, int lane) {
  AFrag f;
  const int m = lane & 15, off = (lane >> 4) * 8;
  const _Float16* p = base + (long)m * stride;
#pragma unroll
  for (int i = 0; i < 8; ++i) { f.e[i] = p[off + i]; f.e[8 + i] = p[16 + off + i]; }
  return f.v;
}

__device__ __forceinline__ v16h load_a_f32(const float* base, int stride, int lane) {
  AFrag f;
  const int m = lane & 15, off = (lane >> 4) * 8;
  const float* p = base + (long)m * stride;
#pragma unroll
  for (int i = 0; i < 8; ++i) {
    f.e[i]     = (_Float16)p[off + i];
    f.e[8 + i] = (_Float16)p[16 + off + i];
  }
  return f.v;
}

// B-matrix 32x16 f16, sourced from a transposed [n][k] array:
// lane L holds column n=L%16, K runs [8*half, 8*half+8) and [16+8*half, 16+8*half+8).
__device__ __forceinline__ v16h load_bt_f16(const _Float16* baseT, int stride, int lane) {
  AFrag f;
  const int n = lane & 15, kb = (lane >> 4) * 8;
  const _Float16* p = baseT + (long)n * stride + kb;
#pragma unroll
  for (int i = 0; i < 8; ++i) { f.e[i] = p[i]; f.e[8 + i] = p[16 + i]; }
  return f.v;
}

__device__ __forceinline__ v16h load_bt_f32(const float* baseT, int stride, int lane) {
  AFrag f;
  const int n = lane & 15, kb = (lane >> 4) * 8;
  const float* p = baseT + (long)n * stride + kb;
#pragma unroll
  for (int i = 0; i < 8; ++i) { f.e[i] = (_Float16)p[i]; f.e[8 + i] = (_Float16)p[16 + i]; }
  return f.v;
}

__device__ __forceinline__ v8f wmma_f16(v16h a, v16h b, v8f c) {
  return __builtin_amdgcn_wmma_f32_16x16x32_f16(false, a, false, b, (short)0, c, false, false);
}

// ---- Problem constants -----------------------------------------------------
#define BATCH 16
#define D     2048
#define DK    64
#define NR    4
#define NCAND 512
#define KW    5

// ---- Kernel 1: layernorm + query conv -> Qh[b][q][64] f16 ------------------
__global__ __launch_bounds__(256) void ln_qconv_kernel(
    const float* __restrict__ feat, const float* __restrict__ g,
    const float* __restrict__ be, const float* __restrict__ Wq,
    _Float16* __restrict__ Qh) {
  const int b = blockIdx.x, tid = threadIdx.x;
  __shared__ float ln[D];
  __shared__ float red[256];
  const float* x = feat + (long)b * D;

  float s = 0.f;
  for (int i = tid; i < D; i += 256) s += x[i];
  red[tid] = s; __syncthreads();
  for (int st = 128; st > 0; st >>= 1) { if (tid < st) red[tid] += red[tid + st]; __syncthreads(); }
  const float mu = red[0] / (float)D; __syncthreads();

  float vs = 0.f;
  for (int i = tid; i < D; i += 256) { float d = x[i] - mu; vs += d * d; }
  red[tid] = vs; __syncthreads();
  for (int st = 128; st > 0; st >>= 1) { if (tid < st) red[tid] += red[tid + st]; __syncthreads(); }
  const float rstd = rsqrtf(red[0] / (float)D + 1e-5f); __syncthreads();

  for (int i = tid; i < D; i += 256) ln[i] = (x[i] - mu) * rstd * g[i] + be[i];
  __syncthreads();

  for (int i = tid; i < D * DK; i += 256) {
    const int q = i >> 6, dk = i & 63;
    float acc = 0.f;
#pragma unroll
    for (int t = 0; t < KW; ++t) {
      const int p = q + t;
      const float xv = (p < D) ? ln[p] : 0.f;   // zero pad to QPAD_LEN
      acc += xv * Wq[dk * KW + t];
    }
    Qh[((long)b * D + q) * DK + dk] = (_Float16)acc;
  }
}

// ---- Kernel 2: gather + key/value conv ------------------------------------
// Kh[b][s][64] f16, Vh[b][v][2048] f16, s = r*512 + c
__global__ __launch_bounds__(256) void kvconv_kernel(
    const float* __restrict__ feat, const int* __restrict__ idx,
    const float* __restrict__ Wk, const float* __restrict__ Wv,
    _Float16* __restrict__ Kh, _Float16* __restrict__ Vh) {
  const int b = blockIdx.x >> 2, r = blockIdx.x & 3, tid = threadIdx.x;
  __shared__ float sf[D + 1];   // SAME_PAD = 2049
  for (int i = tid; i <= D; i += 256)
    sf[i] = (i < D) ? feat[(long)b * D + idx[r * D + i]] : 0.f;
  __syncthreads();

  for (int i = tid; i < NCAND * DK; i += 256) {
    const int c = i >> 6, dk = i & 63;
    float ak = 0.f, av = 0.f;
#pragma unroll
    for (int t = 0; t < KW; ++t) {
      const float xv = sf[4 * c + t];
      ak += xv * Wk[dk * KW + t];
      av += xv * Wv[dk * KW + t];
    }
    const int s = r * NCAND + c;
    Kh[((long)b * D + s) * DK + dk] = (_Float16)ak;
    Vh[((long)b * DK + dk) * D + s] = (_Float16)av;
  }
}

// ---- Kernel 3: FUSED attention --------------------------------------------
// Per (b, 16-row q-tile): S = Q^T K into LDS (16x2048 f32, 128 KB),
// row softmax in LDS (normalized attn written to global exactly once),
// O = P @ V^T from LDS, then Wc contraction + residual -> out_pre.
__global__ __launch_bounds__(256) void fused_attn_kernel(
    const _Float16* __restrict__ Qh, const _Float16* __restrict__ Kh,
    const _Float16* __restrict__ Vh, const float* __restrict__ Wc,
    const float* __restrict__ feat, float* __restrict__ attn,
    float* __restrict__ out_pre) {
  const int b = blockIdx.x >> 7, q0 = (blockIdx.x & 127) * 16;
  const int tid = threadIdx.x, wave = tid >> 5, lane = tid & 31;

  __shared__ float S[16][D];        // 128 KB score strip
  __shared__ float Op[2][16][DK];   // 8 KB partial O

  // ---- Phase A: scores into LDS ----
  {
    const _Float16* Qb = Qh + ((long)b * D + q0) * DK;
    const v16h a0 = load_a_f16(Qb,      DK, lane);   // K chunk 0..31
    const v16h a1 = load_a_f16(Qb + 32, DK, lane);   // K chunk 32..63
    const int n = lane & 15, mh = (lane >> 4) * 8;
    for (int jj = 0; jj < 16; ++jj) {
      const int s0 = (jj * 8 + wave) * 16;
      const _Float16* Kb = Kh + ((long)b * D + s0) * DK;
      CFrag c;
#pragma unroll
      for (int rr = 0; rr < 8; ++rr) c.e[rr] = 0.f;
      c.v = wmma_f16(a0, load_bt_f16(Kb,      DK, lane), c.v);
      c.v = wmma_f16(a1, load_bt_f16(Kb + 32, DK, lane), c.v);
#pragma unroll
      for (int rr = 0; rr < 8; ++rr) S[mh + rr][s0 + n] = c.e[rr];
    }
  }
  __syncthreads();

  // ---- Phase B: row softmax in LDS, write normalized attn to global ----
#pragma unroll
  for (int rr = 0; rr < 2; ++rr) {
    const int row = wave * 2 + rr;
    float4* rowp = (float4*)&S[row][0];          // 512 float4 per row

    float mx = -3.4e38f;
    for (int i = lane; i < D / 4; i += 32) {
      const float4 v = rowp[i];
      mx = fmaxf(mx, fmaxf(fmaxf(v.x, v.y), fmaxf(v.z, v.w)));
    }
#pragma unroll
    for (int o = 16; o > 0; o >>= 1) mx = fmaxf(mx, __shfl_xor(mx, o, 32));

    float sum = 0.f;
    for (int i = lane; i < D / 4; i += 32) {
      float4 v = rowp[i];
      v.x = __expf(v.x - mx); v.y = __expf(v.y - mx);
      v.z = __expf(v.z - mx); v.w = __expf(v.w - mx);
      sum += v.x + v.y + v.z + v.w;
      rowp[i] = v;
    }
#pragma unroll
    for (int o = 16; o > 0; o >>= 1) sum += __shfl_xor(sum, o, 32);
    const float inv = 1.f / sum;

    float4* grow = (float4*)(attn + ((long)b * D + q0 + row) * D);
    for (int i = lane; i < D / 4; i += 32) {
      float4 v = rowp[i];
      v.x *= inv; v.y *= inv; v.z *= inv; v.w *= inv;
      rowp[i] = v;
      grow[i] = v;       // the ONLY global write of attn
    }
  }
  __syncthreads();

  // ---- Phase C: O = P @ V^T (P from LDS, f32->f16 in fragment load) ----
  {
    const int n0 = (wave & 3) * 16;   // v-tile
    const int kh = wave >> 2;         // k-half (chunks 0..31 or 32..63)
    CFrag c;
#pragma unroll
    for (int rr = 0; rr < 8; ++rr) c.e[rr] = 0.f;

    const _Float16* Vb = Vh + ((long)b * DK + n0) * D;
    for (int ch = 0; ch < 32; ++ch) {
      const int koff = (kh * 32 + ch) * 32;
      c.v = wmma_f16(load_a_f32(&S[0][0] + koff, D, lane),
                     load_bt_f16(Vb + koff, D, lane), c.v);
    }
    const int n = lane & 15, mh = (lane >> 4) * 8;
#pragma unroll
    for (int rr = 0; rr < 8; ++rr) Op[kh][mh + rr][n0 + n] = c.e[rr];
  }
  __syncthreads();

  // ---- Phase D: 1x1 conv over v (Wc) + residual ----
  if (tid < 16) {
    float acc = 0.f;
    for (int v = 0; v < DK; ++v) acc += (Op[0][tid][v] + Op[1][tid][v]) * Wc[v];
    out_pre[(long)b * D + q0 + tid] = acc + feat[(long)b * D + q0 + tid];
  }
}

// ---- Kernel 4: bottleneck pre-LN -> h16 f16 --------------------------------
__global__ __launch_bounds__(256) void bln_kernel(
    const float* __restrict__ xin, const float* __restrict__ g,
    const float* __restrict__ be, _Float16* __restrict__ h16) {
  const int b = blockIdx.x, tid = threadIdx.x;
  const float* x = xin + (long)b * D;
  __shared__ float red[256];

  float s = 0.f;
  for (int i = tid; i < D; i += 256) s += x[i];
  red[tid] = s; __syncthreads();
  for (int st = 128; st > 0; st >>= 1) { if (tid < st) red[tid] += red[tid + st]; __syncthreads(); }
  const float mu = red[0] / (float)D; __syncthreads();

  float vs = 0.f;
  for (int i = tid; i < D; i += 256) { float d = x[i] - mu; vs += d * d; }
  red[tid] = vs; __syncthreads();
  for (int st = 128; st > 0; st >>= 1) { if (tid < st) red[tid] += red[tid + st]; __syncthreads(); }
  const float rstd = rsqrtf(red[0] / (float)D + 1e-6f);

  for (int i = tid; i < D; i += 256)
    h16[(long)b * D + i] = (_Float16)((x[i] - mu) * rstd * g[i] + be[i]);
}

// ---- Kernel 5: FFN GEMM  C[m][n] = A[m][:] . W[n][:] + bias[n] -------------
// relu!=0: write f16 to outH (hidden). relu==0: write f32 + residual to outF.
__global__ __launch_bounds__(256) void ffn_gemm_kernel(
    const _Float16* __restrict__ A, const float* __restrict__ W,
    const float* __restrict__ bias, const float* __restrict__ resid,
    float* __restrict__ outF, _Float16* __restrict__ outH, int relu) {
  const int tid = threadIdx.x, wave = tid >> 5, lane = tid & 31;
  const int n0 = blockIdx.x * 128 + wave * 16;
  CFrag c;
#pragma unroll
  for (int rr = 0; rr < 8; ++rr) c.e[rr] = 0.f;

  for (int ch = 0; ch < 64; ++ch) {
    const int koff = ch * 32;
    c.v = wmma_f16(load_a_f16(A + koff, D, lane),
                   load_bt_f32(W + (long)n0 * D + koff, D, lane), c.v);
  }
  const int n = lane & 15, mh = (lane >> 4) * 8;
  const float bv = bias[n0 + n];
#pragma unroll
  for (int rr = 0; rr < 8; ++rr) {
    const int m = mh + rr;
    float val = c.e[rr] + bv;
    if (relu) {
      val = fmaxf(val, 0.f);
      outH[(long)m * D + n0 + n] = (_Float16)val;
    } else {
      outF[(long)m * D + n0 + n] = val + resid[(long)m * D + n0 + n];
    }
  }
}

// ---- Host launcher ---------------------------------------------------------
extern "C" void kernel_launch(void* const* d_in, const int* in_sizes, int n_in,
                              void* d_out, int out_size, void* d_ws, size_t ws_size,
                              hipStream_t stream) {
  (void)in_sizes; (void)n_in; (void)out_size; (void)ws_size;
  const float* feat  = (const float*)d_in[0];
  const int*   idx   = (const int*)  d_in[1];
  const float* Wq    = (const float*)d_in[2];
  const float* Wk    = (const float*)d_in[3];
  const float* Wv    = (const float*)d_in[4];
  const float* Wc    = (const float*)d_in[5];
  const float* ln_g  = (const float*)d_in[6];
  const float* ln_b  = (const float*)d_in[7];
  const float* w1    = (const float*)d_in[8];
  const float* b1    = (const float*)d_in[9];
  const float* w2    = (const float*)d_in[10];
  const float* b2    = (const float*)d_in[11];
  const float* blng  = (const float*)d_in[12];
  const float* blnb  = (const float*)d_in[13];

  float* out  = (float*)d_out;                 // [16][2048]
  float* attn = out + (long)BATCH * D;         // [16][2048][2048]

  char* w = (char*)d_ws;
  _Float16* Qh      = (_Float16*)(w);                       // 4 MB
  _Float16* Kh      = (_Float16*)(w + (size_t)4194304);     // 4 MB
  _Float16* Vh      = (_Float16*)(w + (size_t)8388608);     // 4 MB
  float*    out_pre = (float*)   (w + (size_t)12582912);    // 128 KB
  _Float16* h16     = (_Float16*)(w + (size_t)12713984);    // 64 KB
  _Float16* h1      = (_Float16*)(w + (size_t)12779520);    // 64 KB

  ln_qconv_kernel<<<BATCH, 256, 0, stream>>>(feat, ln_g, ln_b, Wq, Qh);
  kvconv_kernel<<<BATCH * NR, 256, 0, stream>>>(feat, idx, Wk, Wv, Kh, Vh);
  fused_attn_kernel<<<BATCH * 128, 256, 0, stream>>>(Qh, Kh, Vh, Wc, feat, attn, out_pre);
  bln_kernel<<<BATCH, 256, 0, stream>>>(out_pre, blng, blnb, h16);
  ffn_gemm_kernel<<<16, 256, 0, stream>>>(h16, w1, b1, nullptr, nullptr, h1, 1);
  ffn_gemm_kernel<<<16, 256, 0, stream>>>(h1, w2, b2, out_pre, out, nullptr, 0);
}
